// RC_RadLIF_63634235458297
// MI455X (gfx1250) — matrix operand order repaired and checked by reading, not compiled, not run on hardware
//
#include <hip/hip_runtime.h>

typedef __attribute__((ext_vector_type(16))) __bf16 v16bf;
typedef __attribute__((ext_vector_type(8)))  float  v8f;
typedef __attribute__((ext_vector_type(4)))  unsigned int v4u;

#define B_   128
#define T_   1000
#define DIN  700
#define HID_ 128
#define OUTD 20

#define AS_STR 40    // LDS row stride (bf16) for 128x32 A/B tiles: 80 B/row (16B-aligned)
#define HS_STR 136   // LDS row stride (bf16) for 128x128 H tile: 272 B/row (16B-aligned)

// ---------------- helpers ----------------

__device__ __forceinline__ unsigned short f32_to_bf16(float f) {
  unsigned int u = __float_as_uint(f);
  unsigned int r = 0x7FFFu + ((u >> 16) & 1u);   // round-to-nearest-even
  return (unsigned short)((u + r) >> 16);
}

union FragBF { v16bf v; unsigned int d[8]; };

// A-matrix 16x32 bf16 fragment, lane m = l16, half hf:
//   d[0..7] <- K = hf*16 + {0..15} (matches ld_frag_b)
// p0 = &tile[row][0]; row stride must keep p0 16B-aligned.
__device__ __forceinline__ v16bf ld_frag_a(const unsigned short* p0, int hf) {
  FragBF f;
  const v4u lo = *(const v4u*)(p0 + hf * 16);
  const v4u hi = *(const v4u*)(p0 + hf * 16 + 8);
  f.d[0] = lo[0]; f.d[1] = lo[1]; f.d[2] = lo[2]; f.d[3] = lo[3];
  f.d[4] = hi[0]; f.d[5] = hi[1]; f.d[6] = hi[2]; f.d[7] = hi[3];
  return f.v;
}

// B-matrix 32x16 bf16 fragment: lane n = l16, K = hf*16 + e -> 16 consecutive bf16
// p0 = &tile[n][hf*16] (tile stored [n][k]); 16B-aligned.
__device__ __forceinline__ v16bf ld_frag_b(const unsigned short* p0) {
  FragBF f;
  const v4u lo = *(const v4u*)(p0);
  const v4u hi = *(const v4u*)(p0 + 8);
  f.d[0] = lo[0]; f.d[1] = lo[1]; f.d[2] = lo[2]; f.d[3] = lo[3];
  f.d[4] = hi[0]; f.d[5] = hi[1]; f.d[6] = hi[2]; f.d[7] = hi[3];
  return f.v;
}

// pack 16 f32 -> 16 bf16 and store as two b128s (dst 16B-aligned)
__device__ __forceinline__ void st_pk16(unsigned short* dst, const float* src) {
  unsigned int p[8];
#pragma unroll
  for (int j = 0; j < 8; ++j)
    p[j] = (unsigned int)f32_to_bf16(src[2 * j]) |
           ((unsigned int)f32_to_bf16(src[2 * j + 1]) << 16);
  ((v4u*)dst)[0] = (v4u){p[0], p[1], p[2], p[3]};
  ((v4u*)dst)[1] = (v4u){p[4], p[5], p[6], p[7]};
}

// ---------------- kernel 1: A1m = A1_w * mask.T (bf16) + A_norm ----------------

__global__ __launch_bounds__(256) void k_prep(const float* __restrict__ A1_w,
                                              const float* __restrict__ mask,
                                              unsigned short* __restrict__ a1m_bf16,
                                              float* __restrict__ anorm) {
  __shared__ float red[256];
  const int tid = threadIdx.x;
  float s = 0.0f;
  for (int i = tid; i < HID_ * HID_; i += 256) {
    const int g = i >> 7, h = i & 127;
    const float v = A1_w[i] * mask[h * HID_ + g];   // A1m[g][h]
    a1m_bf16[i] = f32_to_bf16(v);
    s += fabsf(v);
  }
  red[tid] = s;
  __syncthreads();
  for (int off = 128; off > 0; off >>= 1) {
    if (tid < off) red[tid] += red[tid + off];
    __syncthreads();
  }
  if (tid == 0) *anorm = red[0];
}

// ---------------- kernel 2: fused X = relu(inp@W_in^T + b_in) @ A1m^T + b_A1 ----------------

__global__ __launch_bounds__(256) void k_gemm(const float* __restrict__ inp,
                                              const float* __restrict__ W_in,
                                              const float* __restrict__ b_in,
                                              const float* __restrict__ b_A1,
                                              const unsigned short* __restrict__ a1m,
                                              float* __restrict__ X) {
  // 34816 B of LDS; A/B 32-wide tiles during GEMM1, H tile (aliased) afterwards
  __shared__ unsigned short smem[128 * HS_STR];
  unsigned short* As = smem;                   // [128][AS_STR] rows = m
  unsigned short* Bs = smem + 128 * AS_STR;    // [128][AS_STR] rows = n (h)
  unsigned short* Hs = smem;                   // [128][HS_STR] rows = m, cols = h

  const int tid  = threadIdx.x;
  const int wave = tid >> 5;
  const int lane = tid & 31;
  const int l16  = lane & 15;
  const int hf   = (lane >> 4) & 1;
  const int wrow = wave * 16;
  const size_t mbase = (size_t)blockIdx.x * 128;

  const int ldrow = tid >> 1;            // tile row loaded by this thread
  const int ldk   = (tid & 1) * 16;      // k sub-range [ldk, ldk+16)
  const float* arow = inp  + (mbase + (size_t)ldrow) * DIN + ldk;
  const float* brow = W_in + (size_t)ldrow * DIN + ldk;

  v8f acc[8];
#pragma unroll
  for (int nb = 0; nb < 8; ++nb)
#pragma unroll
    for (int i = 0; i < 8; ++i) acc[nb][i] = 0.0f;

  // ---- GEMM1: K = 700, 22 steps of 32 (tail zero-padded), software-pipelined ----
  float areg[16], breg[16];
#pragma unroll
  for (int j = 0; j < 16; ++j) {         // preload kk = 0 (ldk+j < 700 always here)
    areg[j] = arow[j];
    breg[j] = brow[j];
  }

  for (int kk = 0; kk < DIN; kk += 32) {
    st_pk16(As + ldrow * AS_STR + ldk, areg);    // pack + 2x ds_store_b128 each
    st_pk16(Bs + ldrow * AS_STR + ldk, breg);
    __syncthreads();

    const int kn = kk + 32;                      // prefetch next tile into regs
    if (kn < DIN) {
#pragma unroll
      for (int j = 0; j < 16; ++j) {
        const int k = kn + ldk + j;
        areg[j] = (k < DIN) ? arow[kn + j] : 0.0f;
        breg[j] = (k < DIN) ? brow[kn + j] : 0.0f;
      }
    }

    const v16bf af = ld_frag_a(As + (wrow + l16) * AS_STR, hf);
#pragma unroll
    for (int nb = 0; nb < 8; ++nb) {
      const v16bf bf = ld_frag_b(Bs + (nb * 16 + l16) * AS_STR + hf * 16);
      acc[nb] = __builtin_amdgcn_wmma_f32_16x16x32_bf16(
          false, af, false, bf, (short)0, acc[nb], false, false);
    }
    __syncthreads();
  }

  // ---- bias + relu -> bf16 H tile in LDS (aliases dead A/B tiles) ----
#pragma unroll
  for (int nb = 0; nb < 8; ++nb) {
    const int col = nb * 16 + l16;
    const float bias = b_in[col];
#pragma unroll
    for (int r = 0; r < 8; ++r) {
      float h = acc[nb][r] + bias;        // C layout: m = r + 8*hf, n = l16
      h = h > 0.0f ? h : 0.0f;
      Hs[(wrow + r + hf * 8) * HS_STR + col] = f32_to_bf16(h);
    }
  }
  __syncthreads();

  // ---- GEMM2: X = H @ A1m^T ; B[k=h][n=g] = A1m[g][h] read from global (L0-resident) ----
#pragma unroll
  for (int nb = 0; nb < 8; ++nb)
#pragma unroll
    for (int i = 0; i < 8; ++i) acc[nb][i] = 0.0f;

#pragma unroll
  for (int ks = 0; ks < HID_; ks += 32) {
    const v16bf af = ld_frag_a(Hs + (wrow + l16) * HS_STR + ks, hf);
#pragma unroll
    for (int nb = 0; nb < 8; ++nb) {
      const v16bf bf = ld_frag_b(a1m + (nb * 16 + l16) * HID_ + ks + hf * 16);
      acc[nb] = __builtin_amdgcn_wmma_f32_16x16x32_bf16(
          false, af, false, bf, (short)0, acc[nb], false, false);
    }
  }

#pragma unroll
  for (int nb = 0; nb < 8; ++nb) {
    const int g = nb * 16 + l16;
    const float bias = b_A1[g];
#pragma unroll
    for (int r = 0; r < 8; ++r) {
      X[(mbase + (size_t)(wrow + r + hf * 8)) * HID_ + g] = acc[nb][r] + bias;
    }
  }
}

// ---------------- kernel 3: RadLIF scan (elementwise over (b,h), sequential in t) ----------------

__global__ __launch_bounds__(256) void k_scan(const float* __restrict__ X,
                                              const float* __restrict__ mem0,
                                              const float* __restrict__ thr,
                                              const float* __restrict__ a,
                                              const float* __restrict__ b,
                                              const float* __restrict__ alpha,
                                              const float* __restrict__ beta,
                                              float* __restrict__ out_mem,
                                              float* __restrict__ out_spk,
                                              float* __restrict__ sum_spk) {
  const int i  = blockIdx.x * blockDim.x + threadIdx.x;  // 0..16383
  const int bb = i >> 7;
  const int h  = i & 127;

  const float al = alpha[h], be = beta[h], av = a[h], bv = b[h], th = thr[h];
  float mem = mem0[i];
  float spk = 0.0f, W = 0.0f, ssum = 0.0f;

  const size_t obase = (size_t)bb * (T_ + 1) * HID_ + h;
  out_mem[obase] = mem;        // prepended t=0 state
  out_spk[obase] = 0.0f;

  const float* xp = X + (size_t)bb * T_ * HID_ + h;
  for (int t = 0; t < T_; ++t) {
    const float x     = xp[(size_t)t * HID_];
    const float mem_n = al * mem + (1.0f - al) * (x - W) - th * spk;
    const float W_n   = be * W + (1.0f - be) * av * mem + bv * spk;
    const float spike = (mem_n - th > 0.0f) ? 1.0f : 0.0f;
    mem = mem_n; W = W_n; spk = spike;
    ssum += spike;
    out_mem[obase + (size_t)(t + 1) * HID_] = mem_n;
    out_spk[obase + (size_t)(t + 1) * HID_] = spike;
  }
  sum_spk[i] = ssum * (1.0f / (float)T_);
}

// ---------------- kernel 4: out = sum1_spk @ W_out^T + b_out ----------------

__global__ __launch_bounds__(256) void k_out(const float* __restrict__ sum_spk,
                                             const float* __restrict__ W_out,
                                             const float* __restrict__ b_out,
                                             float* __restrict__ out) {
  const int i = blockIdx.x * blockDim.x + threadIdx.x;
  if (i >= B_ * OUTD) return;
  const int bb = i / OUTD, o = i % OUTD;
  float s = b_out[o];
#pragma unroll 4
  for (int h = 0; h < HID_; ++h) s += sum_spk[bb * HID_ + h] * W_out[o * HID_ + h];
  out[i] = s;
}

// ---------------- launch ----------------

extern "C" void kernel_launch(void* const* d_in, const int* in_sizes, int n_in,
                              void* d_out, int out_size, void* d_ws, size_t ws_size,
                              hipStream_t stream) {
  const float* input = (const float*)d_in[0];
  const float* mask  = (const float*)d_in[1];
  const float* mem0  = (const float*)d_in[2];
  const float* W_in  = (const float*)d_in[3];
  const float* b_in  = (const float*)d_in[4];
  const float* A1_w  = (const float*)d_in[5];
  const float* b_A1  = (const float*)d_in[6];
  const float* W_out = (const float*)d_in[7];
  const float* b_out = (const float*)d_in[8];
  const float* thr   = (const float*)d_in[9];
  const float* a     = (const float*)d_in[10];
  const float* b     = (const float*)d_in[11];
  const float* alpha = (const float*)d_in[12];
  const float* beta  = (const float*)d_in[13];

  // d_out layout: out[2560] | hid1_mem[128*1001*128] | hid1_spk[same] | A_norm[1]
  const size_t HID1 = (size_t)B_ * (T_ + 1) * HID_;   // 16,400,384
  float* outp    = (float*)d_out;
  float* hid_mem = outp + (size_t)B_ * OUTD;
  float* hid_spk = hid_mem + HID1;
  float* anorm   = hid_spk + HID1;

  // workspace: a1m bf16 (32KB) | X f32 (65.5MB) | sum_spk (64KB)
  char* ws = (char*)d_ws;
  unsigned short* a1m = (unsigned short*)ws;
  float* X    = (float*)(ws + 32768);
  float* ssum = (float*)(ws + 32768 + (size_t)B_ * T_ * HID_ * sizeof(float));

  k_prep<<<1, 256, 0, stream>>>(A1_w, mask, a1m, anorm);
  k_gemm<<<(B_ * T_) / 128, 256, 0, stream>>>(input, W_in, b_in, b_A1, a1m, X);
  k_scan<<<(B_ * HID_) / 256, 256, 0, stream>>>(X, mem0, thr, a, b, alpha, beta,
                                                hid_mem, hid_spk, ssum);
  k_out<<<(B_ * OUTD + 255) / 256, 256, 0, stream>>>(ssum, W_out, b_out, outp);
}
